// Attention_13348758355995
// MI455X (gfx1250) — matrix operand order repaired and checked by reading, not compiled, not run on hardware
//
#include <hip/hip_runtime.h>

typedef __bf16 bf16;
typedef __attribute__((ext_vector_type(8)))  bf16     v8bf;
typedef __attribute__((ext_vector_type(16))) bf16     v16bf;
typedef __attribute__((ext_vector_type(8)))  float    v8f;
typedef __attribute__((ext_vector_type(4)))  unsigned v4u;
typedef __attribute__((ext_vector_type(8)))  unsigned v8u;

constexpr int B_ = 2, S_ = 2048, D_ = 4096, H_ = 32, KVH_ = 8, HD_ = 128;
constexpr int REP_ = H_ / KVH_;
constexpr float SCALE_ = 0.08838834764831845f; // HD^-0.5

static __device__ __forceinline__ v16bf cat8(v8bf lo, v8bf hi) {
  union { v16bf v; v8bf h[2]; } u;
  u.h[0] = lo; u.h[1] = hi;
  return u.v;
}
static __device__ __forceinline__ v8f v8f_zero() {
  v8f z = {0.f, 0.f, 0.f, 0.f, 0.f, 0.f, 0.f, 0.f};
  return z;
}
static __device__ __forceinline__ unsigned rfl(unsigned x) {
  return (unsigned)__builtin_amdgcn_readfirstlane((int)x);
}

// ---------------------------------------------------------------------------
// Tensor Data Mover: 2D tile (tile_w elems per row, tile_h rows, row stride
// in elements of 2 bytes) global -> LDS.  D# per CDNA5 ISA ch.8:
//   group0: [1:0]=count=1, [63:32]=lds_addr, [120:64]=global_addr, [127:126]=2
//   group1: [17:16]=data_size(1=2B), [79:48]=tensor_dim0, [111:80]=tensor_dim1,
//           [127:112]=tile_dim0, [143:128]=tile_dim1, [207:160]=dim0_stride
// ---------------------------------------------------------------------------
static __device__ __forceinline__ void tdm_load_2d(unsigned lds_off,
                                                   unsigned long long ga,
                                                   unsigned tw, unsigned th,
                                                   unsigned long long rs) {
  v4u g0;
  g0[0] = rfl(1u);                                   // count=1, user mode
  g0[1] = rfl(lds_off);                              // lds_addr (bytes)
  g0[2] = rfl((unsigned)ga);                         // global_addr[31:0]
  g0[3] = rfl(((unsigned)(ga >> 32) & 0x01FFFFFFu) | (2u << 30)); // [56:32]|type=2
  v8u g1;
  g1[0] = rfl(0x00010000u);                          // data_size=1 (2 bytes)
  g1[1] = rfl((tw & 0xFFFFu) << 16);                 // tensor_dim0 lo
  g1[2] = rfl((tw >> 16) | ((th & 0xFFFFu) << 16));  // dim0 hi | tensor_dim1 lo
  g1[3] = rfl((th >> 16) | ((tw & 0xFFFFu) << 16));  // dim1 hi | tile_dim0
  g1[4] = rfl(th & 0xFFFFu);                         // tile_dim1 | tile_dim2=0
  g1[5] = rfl((unsigned)rs);                         // dim0_stride[31:0]
  g1[6] = rfl((unsigned)((rs >> 32) & 0xFFFFu));     // dim0_stride[47:32]
  g1[7] = rfl(0u);
  asm volatile("tensor_load_to_lds %0, %1" :: "s"(g0), "s"(g1) : "memory");
}

// ---------------------------------------------------------------------------
// Elementwise conversion: f32 -> bf16
// ---------------------------------------------------------------------------
__global__ void f32_to_bf16_kernel(const float* __restrict__ src,
                                   bf16* __restrict__ dst, size_t n) {
  size_t i = (size_t)blockIdx.x * blockDim.x + threadIdx.x;
  if (i < n) dst[i] = (bf16)src[i];
}

// w is (K,N) f32 row-major; wt is (N,K) bf16 row-major (coalesced writes)
__global__ void transpose_f32_to_bf16_kernel(const float* __restrict__ w,
                                             bf16* __restrict__ wt,
                                             int K, int N) {
  size_t i = (size_t)blockIdx.x * blockDim.x + threadIdx.x;
  size_t total = (size_t)K * N;
  if (i >= total) return;
  int k = (int)(i % K);
  int n = (int)(i / K);
  wt[i] = (bf16)w[(size_t)k * N + n];
}

// ---------------------------------------------------------------------------
// RoPE in place on bf16 tensor of shape (B, S, nh, HD)
// ---------------------------------------------------------------------------
__global__ void rope_kernel(bf16* __restrict__ t, const float* __restrict__ fc,
                            const float* __restrict__ fs, int nh) {
  size_t i = (size_t)blockIdx.x * blockDim.x + threadIdx.x;
  size_t total = (size_t)B_ * S_ * nh * (HD_ / 2);
  if (i >= total) return;
  int j = (int)(i % (HD_ / 2));
  size_t r1 = i / (HD_ / 2);
  int hh = (int)(r1 % nh);
  size_t bs = r1 / nh;            // b*S + s
  int s = (int)(bs % S_);
  bf16* p = t + (bs * nh + hh) * HD_ + 2 * j;
  float c = fc[(size_t)s * (HD_ / 2) + j];
  float sn = fs[(size_t)s * (HD_ / 2) + j];
  float re = (float)p[0], im = (float)p[1];
  p[0] = (bf16)(re * c - im * sn);
  p[1] = (bf16)(re * sn + im * c);
}

// ---------------------------------------------------------------------------
// V (B,S,KVH,HD) -> Vt (B,KVH,HD,S)  so P*V B-fragments are contiguous in k
// ---------------------------------------------------------------------------
__global__ void transpose_v_kernel(const bf16* __restrict__ Vb,
                                   bf16* __restrict__ Vt) {
  size_t i = (size_t)blockIdx.x * blockDim.x + threadIdx.x;
  size_t total = (size_t)B_ * S_ * KVH_ * HD_;
  if (i >= total) return;
  int d = (int)(i % HD_);
  size_t r1 = i / HD_;
  int kvh = (int)(r1 % KVH_);
  size_t r2 = r1 / KVH_;
  int s = (int)(r2 % S_);
  int b = (int)(r2 / S_);
  Vt[(((size_t)b * KVH_ + kvh) * HD_ + d) * S_ + s] = Vb[i];
}

// ---------------------------------------------------------------------------
// GEMM: C[M,N] = A[M,K] * Bt[N,K]^T (bf16, f32 accum).
// Block = 8 waves stacked in M (256 rows) x 64 columns; each wave computes a
// 32x64 tile (2 M-tiles x 4 N-tiles): per 32-wide K step, 12 b128 loads feed
// 8 v_wmma_f32_16x16x32_bf16 (B fragments reused across both M-tiles).
// ---------------------------------------------------------------------------
__global__ __launch_bounds__(256) void gemm_bf16_wmma(
    const bf16* __restrict__ A, const bf16* __restrict__ Bt,
    bf16* __restrict__ Cb, float* __restrict__ Cf,
    int M, int N, int K, int out_f32) {
  const int wid = threadIdx.x >> 5;
  const int lane = threadIdx.x & 31;
  const int m = lane & 15;   // A row / B col / C col within tile
  const int g = lane >> 4;   // half-wave selector
  const int m0 = blockIdx.x * 256 + wid * 32;
  const int n0 = blockIdx.y * 64;

  v8f acc[2][4];
#pragma unroll
  for (int u = 0; u < 2; ++u)
#pragma unroll
    for (int t = 0; t < 4; ++t) acc[u][t] = v8f_zero();

  const bf16* arow0 = A + (size_t)(m0 + m) * K;
  const bf16* arow1 = A + (size_t)(m0 + 16 + m) * K;
  for (int k0 = 0; k0 < K; k0 += 32) {
    // A fragment: lanes 0-15 hold K 0-15; lanes 16-31 hold 16-31
    v16bf af0 = cat8(*(const v8bf*)(arow0 + k0 + g * 16),
                     *(const v8bf*)(arow0 + k0 + g * 16 + 8));
    v16bf af1 = cat8(*(const v8bf*)(arow1 + k0 + g * 16),
                     *(const v8bf*)(arow1 + k0 + g * 16 + 8));
#pragma unroll
    for (int t = 0; t < 4; ++t) {
      // B fragment: lane = column n, 16 consecutive K values (g selects half)
      const bf16* bp = Bt + (size_t)(n0 + t * 16 + m) * K + k0 + g * 16;
      v16bf bfr = *(const v16bf*)bp;
      acc[0][t] = __builtin_amdgcn_wmma_f32_16x16x32_bf16(
          false, af0, false, bfr, (short)0, acc[0][t], false, false);
      acc[1][t] = __builtin_amdgcn_wmma_f32_16x16x32_bf16(
          false, af1, false, bfr, (short)0, acc[1][t], false, false);
    }
  }
  // C layout: lane col = n0+t*16+m, rows r (lanes 0-15) / r+8 (lanes 16-31)
#pragma unroll
  for (int u = 0; u < 2; ++u) {
#pragma unroll
    for (int t = 0; t < 4; ++t) {
#pragma unroll
      for (int r = 0; r < 8; ++r) {
        const int row = m0 + u * 16 + r + g * 8;
        const int col = n0 + t * 16 + m;
        if (out_f32) Cf[(size_t)row * N + col] = acc[u][t][r];
        else         Cb[(size_t)row * N + col] = (bf16)acc[u][t][r];
      }
    }
  }
}

// ---------------------------------------------------------------------------
// Flash attention, TDM-staged.  8 waves/block = 8 consecutive 16-row Q tiles
// of the same (b,h): wave 0 DMAs each 32-key K tile (32x128) and V tile
// (128x32, from V^T) into double-buffered LDS with tensor_load_to_lds; all
// waves consume via ds_load_b128 fragments.  Causal mask analytic; waves past
// their causal range skip compute but keep the barriers.
// ---------------------------------------------------------------------------
__global__ __launch_bounds__(256) void attn_fa_kernel(
    const bf16* __restrict__ Q, const bf16* __restrict__ Kc,
    const bf16* __restrict__ Vt, bf16* __restrict__ Out) {
  __shared__ __align__(32) bf16 Klds[2][32][128];  // 16 KB
  __shared__ __align__(32) bf16 Vlds[2][128][32];  // 16 KB
  __shared__ __align__(32) bf16 plds[8][16][32];   //  8 KB (per-wave P tile)

  const int wid = threadIdx.x >> 5;
  const int lane = threadIdx.x & 31;
  const int m = lane & 15;
  const int g = lane >> 4;

  const int qtiles = S_ / 16;                    // 128 (16 blocks per (b,h))
  const int tile = blockIdx.x * 8 + wid;
  const int b = tile / (H_ * qtiles);
  const int r0 = tile % (H_ * qtiles);
  const int h = r0 / qtiles;
  const int q0 = (r0 % qtiles) * 16;             // this wave's Q rows
  const int kvh = h / REP_;
  const int q0blk = ((blockIdx.x * 8) % qtiles) * 16;  // block's first Q row
  const int nch = (q0blk + 128) / 32;            // chunks for the whole block

  // Q A-fragments: 4 chunks of 32 over HD=128, kept in VGPRs for whole loop
  const bf16* qb = Q + ((size_t)((size_t)b * S_ + q0) * H_ + h) * HD_;
  v16bf qf[4];
#pragma unroll
  for (int c = 0; c < 4; ++c) {
    const bf16* p = qb + (size_t)m * (H_ * HD_) + c * 32 + g * 16;
    qf[c] = cat8(*(const v8bf*)p, *(const v8bf*)(p + 8));
  }

  v8f o[8];
#pragma unroll
  for (int t = 0; t < 8; ++t) o[t] = v8f_zero();
  float mi[8], li[8];
#pragma unroll
  for (int r = 0; r < 8; ++r) { mi[r] = -3.0e38f; li[r] = 0.0f; }

  const unsigned long long kbg =
      (unsigned long long)(Kc + ((size_t)((size_t)b * S_) * KVH_ + kvh) * HD_);
  const unsigned long long vbg =
      (unsigned long long)(Vt + ((size_t)b * KVH_ + kvh) * HD_ * (size_t)S_);
  const unsigned kl0 = (unsigned)(unsigned long long)(uintptr_t)&Klds[0][0][0];
  const unsigned kl1 = (unsigned)(unsigned long long)(uintptr_t)&Klds[1][0][0];
  const unsigned vl0 = (unsigned)(unsigned long long)(uintptr_t)&Vlds[0][0][0];
  const unsigned vl1 = (unsigned)(unsigned long long)(uintptr_t)&Vlds[1][0][0];

  // ---- prologue: stage chunks 0 and 1 (nch >= 4 always)
  if (wid == 0) {
    tdm_load_2d(kl0, kbg, 128u, 32u, (unsigned long long)(KVH_ * HD_));
    tdm_load_2d(vl0, vbg, 32u, 128u, (unsigned long long)S_);
    tdm_load_2d(kl1, kbg + (unsigned long long)32 * KVH_ * HD_ * 2, 128u, 32u,
                (unsigned long long)(KVH_ * HD_));
    tdm_load_2d(vl1, vbg + 64ull, 32u, 128u, (unsigned long long)S_);
    __builtin_amdgcn_s_wait_tensorcnt(2);   // chunk 0 resident
  }
  __syncthreads();

  for (int ch = 0; ch < nch; ++ch) {
    const int k0 = ch * 32;
    const int buf = ch & 1;
    if (k0 <= q0 + 15) {  // wave-uniform: chunk intersects this wave's range
      // ---- scores: two 16x16 tiles over 32 keys, K-dim = HD via 4 WMMAs
      v8f s0 = v8f_zero(), s1 = v8f_zero();
#pragma unroll
      for (int c = 0; c < 4; ++c) {
        v16bf kf0 = *(const v16bf*)&Klds[buf][m][c * 32 + g * 16];
        v16bf kf1 = *(const v16bf*)&Klds[buf][16 + m][c * 32 + g * 16];
        s0 = __builtin_amdgcn_wmma_f32_16x16x32_bf16(false, qf[c], false, kf0,
                                                     (short)0, s0, false, false);
        s1 = __builtin_amdgcn_wmma_f32_16x16x32_bf16(false, qf[c], false, kf1,
                                                     (short)0, s1, false, false);
      }
      // ---- online softmax (row r lives in a 16-lane half-wave group)
#pragma unroll
      for (int r = 0; r < 8; ++r) {
        const int qrow = q0 + r + g * 8;
        float a = s0[r] * SCALE_ + ((k0 + m > qrow) ? -1.0e9f : 0.0f);
        float bb = s1[r] * SCALE_ + ((k0 + 16 + m > qrow) ? -1.0e9f : 0.0f);
        float mx = fmaxf(a, bb);
        mx = fmaxf(mx, __shfl_xor(mx, 1, 32));
        mx = fmaxf(mx, __shfl_xor(mx, 2, 32));
        mx = fmaxf(mx, __shfl_xor(mx, 4, 32));
        mx = fmaxf(mx, __shfl_xor(mx, 8, 32));
        float mn = fmaxf(mi[r], mx);
        float alpha = __expf(mi[r] - mn);
        float p0 = __expf(a - mn);
        float p1 = __expf(bb - mn);
        float rs = p0 + p1;
        rs += __shfl_xor(rs, 1, 32);
        rs += __shfl_xor(rs, 2, 32);
        rs += __shfl_xor(rs, 4, 32);
        rs += __shfl_xor(rs, 8, 32);
        li[r] = li[r] * alpha + rs;
        mi[r] = mn;
        const int row = r + g * 8;
        plds[wid][row][m] = (bf16)p0;          // C-frag -> A-frag relayout
        plds[wid][row][16 + m] = (bf16)p1;
#pragma unroll
        for (int t = 0; t < 8; ++t) o[t][r] *= alpha;
      }
      v16bf pf = cat8(*(const v8bf*)&plds[wid][m][g * 16],
                      *(const v8bf*)&plds[wid][m][g * 16 + 8]);
      // ---- O += P x V  (8 d-tiles of 16, K-dim = 32 keys)
#pragma unroll
      for (int t = 0; t < 8; ++t) {
        v16bf vf = *(const v16bf*)&Vlds[buf][t * 16 + m][g * 16];
        o[t] = __builtin_amdgcn_wmma_f32_16x16x32_bf16(false, pf, false, vf,
                                                       (short)0, o[t], false, false);
      }
    }
    __syncthreads();  // everyone done reading buf before it is refilled
    if (wid == 0) {
      if (ch + 2 < nch) {
        const int kn = (ch + 2) * 32;
        const unsigned kl = buf ? kl1 : kl0;
        const unsigned vl = buf ? vl1 : vl0;
        tdm_load_2d(kl, kbg + (unsigned long long)kn * KVH_ * HD_ * 2, 128u,
                    32u, (unsigned long long)(KVH_ * HD_));
        tdm_load_2d(vl, vbg + (unsigned long long)kn * 2, 32u, 128u,
                    (unsigned long long)S_);
        __builtin_amdgcn_s_wait_tensorcnt(2);  // chunk ch+1 resident
      } else {
        __builtin_amdgcn_s_wait_tensorcnt(0);
      }
    }
    __syncthreads();  // publish: next buffer ready for all waves
  }

  // ---- normalize and store (B,S,H*HD) bf16
  bf16* ob = Out + ((size_t)((size_t)b * S_ + q0) * H_ + h) * HD_;
#pragma unroll
  for (int r = 0; r < 8; ++r) {
    const float inv = 1.0f / li[r];
    const int row = r + g * 8;
#pragma unroll
    for (int t = 0; t < 8; ++t) {
      ob[(size_t)row * (H_ * HD_) + t * 16 + m] = (bf16)(o[t][r] * inv);
    }
  }
}

// ---------------------------------------------------------------------------
// Host-side orchestration
// ---------------------------------------------------------------------------
extern "C" void kernel_launch(void* const* d_in, const int* in_sizes, int n_in,
                              void* d_out, int out_size, void* d_ws,
                              size_t ws_size, hipStream_t stream) {
  (void)in_sizes; (void)n_in; (void)out_size; (void)ws_size;
  const float* x  = (const float*)d_in[0];
  const float* fc = (const float*)d_in[1];
  const float* fs = (const float*)d_in[2];
  // d_in[3] (mask) unused: causal mask computed analytically in-kernel
  const float* wq = (const float*)d_in[4];
  const float* wk = (const float*)d_in[5];
  const float* wv = (const float*)d_in[6];
  const float* wo = (const float*)d_in[7];
  // d_in[8] (positions) unused: equals arange(S)
  float* out = (float*)d_out;

  const size_t M   = (size_t)B_ * S_;   // 4096 tokens
  const size_t K   = D_;                // 4096
  const size_t NQ  = (size_t)H_ * HD_;  // 4096
  const size_t NKV = (size_t)KVH_ * HD_;// 1024

  bf16* ws  = (bf16*)d_ws;
  bf16* xb  = ws;  ws += M * K;           // x in bf16
  bf16* wqT = ws;  ws += NQ * K;          // wq^T (N,K)
  bf16* wkT = ws;  ws += NKV * K;         // wk^T
  bf16* wvT = ws;  ws += NKV * K;         // wv^T
  bf16* woT = ws;  ws += (size_t)D_ * NQ; // wo^T (D, H*HD)
  bf16* Qb  = ws;  ws += M * NQ;          // Q  (B,S,H,HD)
  bf16* Kb  = ws;  ws += M * NKV;         // K  (B,S,KVH,HD)
  bf16* Vb  = ws;  ws += M * NKV;         // V  (B,S,KVH,HD)
  bf16* Vt  = ws;  ws += M * NKV;         // V^T (B,KVH,HD,S)
  bf16* attn = xb;                        // reuse x buffer for attention out

  const int TB = 256;
  f32_to_bf16_kernel<<<(int)((M * K + TB - 1) / TB), TB, 0, stream>>>(x, xb, M * K);
  transpose_f32_to_bf16_kernel<<<(int)((K * NQ + TB - 1) / TB), TB, 0, stream>>>(
      wq, wqT, (int)K, (int)NQ);
  transpose_f32_to_bf16_kernel<<<(int)((K * NKV + TB - 1) / TB), TB, 0, stream>>>(
      wk, wkT, (int)K, (int)NKV);
  transpose_f32_to_bf16_kernel<<<(int)((K * NKV + TB - 1) / TB), TB, 0, stream>>>(
      wv, wvT, (int)K, (int)NKV);
  transpose_f32_to_bf16_kernel<<<(int)((NQ * (size_t)D_ + TB - 1) / TB), TB, 0, stream>>>(
      wo, woT, (int)NQ, (int)D_);

  dim3 blk(TB);
  gemm_bf16_wmma<<<dim3((unsigned)(M / 256), (unsigned)(NQ / 64)), blk, 0, stream>>>(
      xb, wqT, Qb, nullptr, (int)M, (int)NQ, (int)K, 0);
  gemm_bf16_wmma<<<dim3((unsigned)(M / 256), (unsigned)(NKV / 64)), blk, 0, stream>>>(
      xb, wkT, Kb, nullptr, (int)M, (int)NKV, (int)K, 0);
  gemm_bf16_wmma<<<dim3((unsigned)(M / 256), (unsigned)(NKV / 64)), blk, 0, stream>>>(
      xb, wvT, Vb, nullptr, (int)M, (int)NKV, (int)K, 0);

  rope_kernel<<<(int)(((size_t)B_ * S_ * H_ * (HD_ / 2) + TB - 1) / TB), TB, 0,
                stream>>>(Qb, fc, fs, H_);
  rope_kernel<<<(int)(((size_t)B_ * S_ * KVH_ * (HD_ / 2) + TB - 1) / TB), TB, 0,
                stream>>>(Kb, fc, fs, KVH_);
  transpose_v_kernel<<<(int)((M * NKV + TB - 1) / TB), TB, 0, stream>>>(Vb, Vt);

  attn_fa_kernel<<<dim3((unsigned)((B_ * H_ * (S_ / 16)) / 8)), blk, 0, stream>>>(
      Qb, Kb, Vt, attn);

  gemm_bf16_wmma<<<dim3((unsigned)(M / 256), (unsigned)(D_ / 64)), blk, 0, stream>>>(
      attn, woT, nullptr, out, (int)M, (int)D_, (int)NQ, 1);
}
